// Attention_10849087389794
// MI455X (gfx1250) — matrix operand order repaired and checked by reading, not compiled, not run on hardware
//
#include <hip/hip_runtime.h>

typedef _Float16 v16h __attribute__((ext_vector_type(16)));
typedef _Float16 v8h  __attribute__((ext_vector_type(8)));
typedef _Float16 v4h  __attribute__((ext_vector_type(4)));
typedef float    v8f  __attribute__((ext_vector_type(8)));
typedef float    v4f  __attribute__((ext_vector_type(4)));

#define B_   16
#define L_   2048
#define D_   128
#define ROWS 32                 // query rows per workgroup
#define KVS  64                 // kv rows per strip
#define NSTRIP (L_ / KVS)       // 32
#define KROW (D_ + 8)           // padded K-strip row (halfs)
#define VROW (KVS + 8)          // padded Vt row (halfs)
#define STAGE_HALFS (D_ * VROW) // 9216 >= KVS*KROW (8704)
#define NTHREADS 256            // 8 waves (wave32)

#define WMMA_F16(A, Bm, C) \
  __builtin_amdgcn_wmma_f32_16x16x32_f16(false, (A), false, (Bm), (short)0, (C), false, false)

#define SHUF16(lo, hi) \
  __builtin_shufflevector((lo), (hi), 0,1,2,3,4,5,6,7,8,9,10,11,12,13,14,15)

__launch_bounds__(NTHREADS)
__global__ void attn_fused_kernel(const float* __restrict__ Q,
                                  const float* __restrict__ K,
                                  const float* __restrict__ V,
                                  float* __restrict__ P,   // [B, L, L]
                                  float* __restrict__ O)   // [B, L, D]
{
    // ---- LDS: 300.5 KB of the 320 KB WGP budget ----
    __shared__ float     S[ROWS][L_];             // 256 KB score/prob rows
    __shared__ _Float16  Qh[ROWS][KROW];          // Q tile, f16, scale folded
    __shared__ _Float16  stage[2][STAGE_HALFS];   // double-buffered K-strip / Vt-strip

    const int tid  = threadIdx.x;
    const int lane = tid & 31;
    const int wave = tid >> 5;                    // 0..7
    const int bid  = blockIdx.x;
    const int b    = bid / (L_ / ROWS);
    const int q0   = (bid % (L_ / ROWS)) * ROWS;

    const float scale = 0.08838834764831845f;     // 1/sqrt(128)

    const float* Qg = Q + ((size_t)b * L_ + q0) * D_;
    const float* Kg = K + (size_t)b * L_ * D_;
    const float* Vg = V + (size_t)b * L_ * D_;

    // wave -> tile assignment (uniform per wave; EXEC all-ones at WMMA)
    const int rb   = wave & 1;                    // query row-block (0/1)
    const int kb   = wave >> 1;                   // kv col-block within strip (0..3)
    const int m0   = rb * 16;
    const int nl   = lane & 15;
    const int hsel = (lane < 16) ? 0 : 16;        // A-frag lane K-offset
    const int bsel = (lane < 16) ? 0 : 16;        // B-frag lane K-offset

    // ============ Phase 0: Q tile -> f16 LDS (scale folded) + preload K0 ====
    #pragma unroll
    for (int j = 0; j < 4; ++j) {                 // rows wave, wave+8, .. (32 rows)
        const int r = wave + j * 8;
        v4f q = *(const v4f*)(Qg + (size_t)r * D_ + lane * 4);
        v4h h;
        h[0] = (_Float16)(q[0] * scale); h[1] = (_Float16)(q[1] * scale);
        h[2] = (_Float16)(q[2] * scale); h[3] = (_Float16)(q[3] * scale);
        *(v4h*)&Qh[r][lane * 4] = h;
    }
    v4f kreg[8];                                  // register-staged strip (64 rows)
    #pragma unroll
    for (int j = 0; j < 8; ++j)
        kreg[j] = *(const v4f*)(Kg + (size_t)(wave + j * 8) * D_ + lane * 4);
    __syncthreads();

    // hoist the wave's 4 Q A-fragments (strip-invariant)
    v16h aq[4];
    #pragma unroll
    for (int kc = 0; kc < 4; ++kc) {
        const _Float16* arow = &Qh[m0 + nl][kc * 32];
        v8h alo = *(const v8h*)(arow + hsel);
        v8h ahi = *(const v8h*)(arow + 8 + hsel);
        aq[kc] = SHUF16(alo, ahi);
    }

    // ============ Phase 1: S = (Q*scale) @ K^T  (1 barrier / strip) ========
    for (int s = 0; s < NSTRIP; ++s) {
        _Float16* Kh = &stage[s & 1][0];
        #pragma unroll
        for (int j = 0; j < 8; ++j) {             // cvt + store staged strip
            v4f k4 = kreg[j];
            v4h h;
            h[0] = (_Float16)k4[0]; h[1] = (_Float16)k4[1];
            h[2] = (_Float16)k4[2]; h[3] = (_Float16)k4[3];
            *(v4h*)&Kh[(wave + j * 8) * KROW + lane * 4] = h;
        }
        __syncthreads();
        if (s + 1 < NSTRIP) {                     // overlap next strip with WMMA
            const float* Ks = Kg + (size_t)((s + 1) * KVS) * D_;
            #pragma unroll
            for (int j = 0; j < 8; ++j)
                kreg[j] = *(const v4f*)(Ks + (size_t)(wave + j * 8) * D_ + lane * 4);
        }
        v8f c = {};
        #pragma unroll
        for (int kc = 0; kc < 4; ++kc) {          // D=128 in K-chunks of 32
            // B-frag: B[k][n] = K[kv=n][d=k] -> contiguous run in Kh row
            const _Float16* brow = &Kh[(kb * 16 + nl) * KROW + kc * 32 + bsel];
            v8h blo = *(const v8h*)(brow);
            v8h bhi = *(const v8h*)(brow + 8);
            c = WMMA_F16(aq[kc], SHUF16(blo, bhi), c);
        }
        // scatter 16x16 f32 tile into S (C layout: VGPR i -> M=i / M=8+i)
        const int col0 = s * KVS + kb * 16;
        #pragma unroll
        for (int i = 0; i < 8; ++i)
            S[m0 + ((lane < 16) ? i : 8 + i)][col0 + nl] = c[i];
    }
    __syncthreads();

    // ============ Phase 2: row softmax, write P (NT stores) ================
    for (int rr = 0; rr < 4; ++rr) {              // wave owns 4 rows
        const int row = wave * 4 + rr;
        v4f* Sr = (v4f*)&S[row][0];
        float m = -1e30f;
        #pragma unroll
        for (int i = 0; i < 16; ++i) {
            v4f x = Sr[i * 32 + lane];
            m = fmaxf(m, fmaxf(fmaxf(x[0], x[1]), fmaxf(x[2], x[3])));
        }
        #pragma unroll
        for (int off = 16; off; off >>= 1) m = fmaxf(m, __shfl_xor(m, off));

        float sum = 0.f;
        #pragma unroll
        for (int i = 0; i < 16; ++i) {
            v4f x = Sr[i * 32 + lane], e;
            e[0] = __expf(x[0] - m); e[1] = __expf(x[1] - m);
            e[2] = __expf(x[2] - m); e[3] = __expf(x[3] - m);
            Sr[i * 32 + lane] = e;
            sum += e[0] + e[1] + e[2] + e[3];
        }
        #pragma unroll
        for (int off = 16; off; off >>= 1) sum += __shfl_xor(sum, off);
        const float inv = 1.0f / sum;

        v4f* Pg = (v4f*)(P + ((size_t)b * L_ + q0 + row) * L_);
        #pragma unroll
        for (int i = 0; i < 16; ++i) {
            v4f e = Sr[i * 32 + lane], p;
            p[0] = e[0] * inv; p[1] = e[1] * inv;
            p[2] = e[2] * inv; p[3] = e[3] * inv;
            Sr[i * 32 + lane] = p;                        // keep probs for P@V
            __builtin_nontemporal_store(p, &Pg[i * 32 + lane]); // 268MB: don't pollute L2
        }
    }
    v4f vreg[8];                                  // preload V strip 0
    #pragma unroll
    for (int j = 0; j < 8; ++j)
        vreg[j] = *(const v4f*)(Vg + (size_t)(wave + j * 8) * D_ + lane * 4);
    __syncthreads();

    // ============ Phase 3: O = P @ V  (1 barrier / strip) ==================
    const int cb0 = wave >> 1;                    // wave owns d-tiles cb0, cb0+4
    v8f acc0 = {}, acc1 = {};
    for (int s = 0; s < NSTRIP; ++s) {
        _Float16* Vt = &stage[s & 1][0];          // transposed: Vt[d][kv]
        #pragma unroll
        for (int j = 0; j < 8; ++j) {
            v4f v4 = vreg[j];
            const int r = wave + j * 8;
            Vt[(lane * 4 + 0) * VROW + r] = (_Float16)v4[0];
            Vt[(lane * 4 + 1) * VROW + r] = (_Float16)v4[1];
            Vt[(lane * 4 + 2) * VROW + r] = (_Float16)v4[2];
            Vt[(lane * 4 + 3) * VROW + r] = (_Float16)v4[3];
        }
        __syncthreads();
        if (s + 1 < NSTRIP) {
            const float* Vs = Vg + (size_t)((s + 1) * KVS) * D_;
            #pragma unroll
            for (int j = 0; j < 8; ++j)
                vreg[j] = *(const v4f*)(Vs + (size_t)(wave + j * 8) * D_ + lane * 4);
        }
        #pragma unroll
        for (int kc = 0; kc < 2; ++kc) {          // strip K=64 in chunks of 32
            const int kvoff = kc * 32;
            // A-frag: probs row (f32 in LDS) -> f16 on the fly
            const float* arow = &S[m0 + nl][s * KVS + kvoff];
            v8f flo = *(const v8f*)(arow + hsel);
            v8f fhi = *(const v8f*)(arow + 8 + hsel);
            v16h a;
            #pragma unroll
            for (int j = 0; j < 8; ++j) {
                a[j]     = (_Float16)flo[j];
                a[8 + j] = (_Float16)fhi[j];
            }
            // B-frag: B[k][n] = V[kv=k][d=n] -> contiguous run in Vt[d] row
            const _Float16* br0 = &Vt[(cb0 * 16 + nl) * VROW + kvoff + bsel];
            v8h b0lo = *(const v8h*)(br0);
            v8h b0hi = *(const v8h*)(br0 + 8);
            acc0 = WMMA_F16(a, SHUF16(b0lo, b0hi), acc0);

            const _Float16* br1 = &Vt[((cb0 + 4) * 16 + nl) * VROW + kvoff + bsel];
            v8h b1lo = *(const v8h*)(br1);
            v8h b1hi = *(const v8h*)(br1 + 8);
            acc1 = WMMA_F16(a, SHUF16(b1lo, b1hi), acc1);
        }
    }

    // write O (two 64B row segments per VGPR across the wave)
    #pragma unroll
    for (int i = 0; i < 8; ++i) {
        const int row = m0 + ((lane < 16) ? i : 8 + i);
        float* orow = O + ((size_t)b * L_ + q0 + row) * D_;
        __builtin_nontemporal_store(acc0[i], &orow[cb0 * 16 + nl]);
        __builtin_nontemporal_store(acc1[i], &orow[(cb0 + 4) * 16 + nl]);
    }
}

extern "C" void kernel_launch(void* const* d_in, const int* in_sizes, int n_in,
                              void* d_out, int out_size, void* d_ws, size_t ws_size,
                              hipStream_t stream) {
    const float* Q = (const float*)d_in[0];
    const float* K = (const float*)d_in[1];
    const float* V = (const float*)d_in[2];
    float* P = (float*)d_out;                     // attention weights
    float* O = P + (size_t)B_ * L_ * L_;          // context vector
    dim3 grid(B_ * (L_ / ROWS));                  // 1024 workgroups
    attn_fused_kernel<<<grid, NTHREADS, 0, stream>>>(Q, K, V, P, O);
}
